// rnn_mp_gat_44495861187266
// MI455X (gfx1250) — compile-verified, matrix operand order checked by reading, not run on hardware
//
#include <hip/hip_runtime.h>
#include <hip/hip_bf16.h>

// ---------------------------------------------------------------------------
// Problem constants (from reference)
// ---------------------------------------------------------------------------
constexpr int NN = 50000;   // nodes
constexpr int NE = 400000;  // edges

// d_out layout (floats): out | nh | nc | eh | ec
constexpr int O_OUT = 0;
constexpr int O_NH  = 200000;
constexpr int O_NC  = 3400000;
constexpr int O_EH  = 6600000;
constexpr int O_EC  = 19400000;

// packed f16 weights in ws (byte offsets), fragment-swizzled order
constexpr size_t WP_EDGE = 0;        // K=64  x N=128
constexpr size_t WP_NODE = 16384;    // K=96  x N=256
constexpr size_t WP_GAT  = 65536;    // K=96  x N=256
constexpr size_t WP_EM1  = 114688;   // K=160 x N=64
constexpr size_t WP_EM2  = 135168;   // K=64  x N=32
constexpr size_t WP_NM1  = 139264;   // K=96  x N=64
constexpr size_t WP_NM2  = 151552;   // K=64  x N=16
constexpr size_t FWS_OFF = 163840;   // f32 scratch base (bytes)

// f32 scratch layout (float offsets from FWS_OFF)
constexpr int F_ENC = 0;         // N*32
constexpr int F_CNT = 1600000;   // N
constexpr int F_AS  = 1650000;   // N*4
constexpr int F_AD  = 1850000;   // N*4
constexpr int F_MM  = 2050000;   // N*4 (init -inf)
constexpr int F_DEN = 2250000;   // N*4
constexpr int F_XH  = 2450000;   // N*256
constexpr int F_GS  = 15250000;  // N*256
constexpr int F_XG  = 28050000;  // N*64
constexpr int F_AGG = 31250000;  // N*32

// ---------------------------------------------------------------------------
// WMMA helpers (wave32, v_wmma_f32_16x16x32_f16)
// ---------------------------------------------------------------------------
typedef __attribute__((ext_vector_type(16))) _Float16 v16h;
typedef __attribute__((ext_vector_type(8)))  _Float16 v8h;
typedef __attribute__((ext_vector_type(8)))  float    v8f;

__device__ __forceinline__ v8f wmma32(v16h a, v16h b, v8f c) {
  return __builtin_amdgcn_wmma_f32_16x16x32_f16(false, a, false, b, (short)0, c,
                                                false, false);
}

// A fragment from row-major f16 tile in LDS (row stride K halves, K%32==0).
// lane L<16: row L, K kbase+{0..7,16..23}; lane L>=16: row L-16, +8.
__device__ __forceinline__ v16h load_a(const _Float16* st, int K, int lane,
                                       int kbase) {
  const int m  = lane & 15;
  const int kh = (lane >> 4) * 8;
  const v8h lo = *(const v8h*)(st + m * K + kbase + kh);
  const v8h hi = *(const v8h*)(st + m * K + kbase + 16 + kh);
  return __builtin_shufflevector(lo, hi, 0, 1, 2, 3, 4, 5, 6, 7, 8, 9, 10, 11,
                                 12, 13, 14, 15);
}

// B fragment from fragment-swizzled LDS: frag (kt,t) lane l is 16 contiguous
// halves at ((kt*Nt+t)*32+l)*16.
__device__ __forceinline__ v16h load_b(const _Float16* sW, int Nt, int kt,
                                       int t, int lane) {
  return *(const v16h*)(sW + (((((kt * Nt) + t) << 5) + lane) << 4));
}

__device__ __forceinline__ v8h cvt8(float4 a, float4 b) {
  v8h r;
  r[0] = (_Float16)a.x; r[1] = (_Float16)a.y;
  r[2] = (_Float16)a.z; r[3] = (_Float16)a.w;
  r[4] = (_Float16)b.x; r[5] = (_Float16)b.y;
  r[6] = (_Float16)b.z; r[7] = (_Float16)b.w;
  return r;
}

__device__ __forceinline__ float sigmf_(float x) { return 1.f / (1.f + __expf(-x)); }
__device__ __forceinline__ float lrelu_(float x) { return x > 0.f ? x : 0.2f * x; }
__device__ __forceinline__ void atomic_max_f(float* a, float v) {
  if (v >= 0.f) atomicMax((int*)a, __float_as_int(v));
  else          atomicMin((unsigned int*)a, __float_as_uint(v));
}

// fragment-swizzle index decode: i -> (k, n) for a K x N weight with Nt ntiles
__device__ __forceinline__ void swz_kn(int i, int Nt, int& k, int& n) {
  const int e    = i & 15;
  const int lane = (i >> 4) & 31;
  const int f    = i >> 9;
  const int t    = f % Nt;
  const int kt   = f / Nt;
  const int kh   = (lane >> 4) * 8;
  n = (lane & 15) + (t << 4);
  k = (kt << 5) + ((e < 8) ? (kh + e) : (16 + kh + (e - 8)));
}

// ---------------------------------------------------------------------------
// Weight packing (f32 -> fragment-swizzled f16)
// ---------------------------------------------------------------------------
__global__ void pack_weights_kernel(const float* __restrict__ nWih, const float* __restrict__ nWhh,
                                    const float* __restrict__ eWih, const float* __restrict__ eWhh,
                                    const float* __restrict__ gatW, const float* __restrict__ emW1,
                                    const float* __restrict__ emW2, const float* __restrict__ nmW1,
                                    const float* __restrict__ nmW2, char* __restrict__ ws) {
  const int gid = blockIdx.x * 256 + threadIdx.x;
  int k, n;
  if (gid < 8192) {                       // edge LSTM: k[0..31]=h, [32,33]=attr
    swz_kn(gid, 8, k, n);
    float v = (k < 32) ? eWhh[n * 32 + k]
                       : ((k < 34) ? eWih[n * 2 + (k - 32)] : 0.f);
    ((_Float16*)(ws + WP_EDGE))[gid] = (_Float16)v;
  } else if (gid < 32768) {               // node LSTM: k[0..63]=h, [64..68]=x
    const int i = gid - 8192;
    swz_kn(i, 16, k, n);
    float v = (k < 64) ? nWhh[n * 64 + k]
                       : ((k < 69) ? nWih[n * 5 + (k - 64)] : 0.f);
    ((_Float16*)(ws + WP_NODE))[i] = (_Float16)v;
  } else if (gid < 57344) {               // GAT: k[0..63]=nh, [64..95]=enc
    const int i = gid - 32768;
    swz_kn(i, 16, k, n);
    ((_Float16*)(ws + WP_GAT))[i] = (_Float16)gatW[n * 96 + k];
  } else if (gid < 67584) {               // edge MLP W1 (K=160)
    const int i = gid - 57344;
    swz_kn(i, 4, k, n);
    ((_Float16*)(ws + WP_EM1))[i] = (_Float16)((k < 130) ? emW1[n * 130 + k] : 0.f);
  } else if (gid < 69632) {               // edge MLP W2 (64x32)
    const int i = gid - 67584;
    swz_kn(i, 2, k, n);
    ((_Float16*)(ws + WP_EM2))[i] = (_Float16)emW2[n * 64 + k];
  } else if (gid < 75776) {               // node MLP W1 (96x64)
    const int i = gid - 69632;
    swz_kn(i, 4, k, n);
    ((_Float16*)(ws + WP_NM1))[i] = (_Float16)nmW1[n * 96 + k];
  } else if (gid < 76800) {               // node MLP W2 (64x16, OUT=4 padded)
    const int i = gid - 75776;
    swz_kn(i, 1, k, n);
    ((_Float16*)(ws + WP_NM2))[i] = (_Float16)((n < 4) ? nmW2[n * 64 + k] : 0.f);
  }
}

__global__ void init_ws_kernel(float* __restrict__ fws) {
  const int gid = blockIdx.x * 256 + threadIdx.x;
  if (gid < 2450000) {
    fws[gid] = (gid >= F_MM && gid < F_DEN) ? -__builtin_inff() : 0.f;
  } else if (gid < 15250000) {
    fws[F_GS + (gid - 2450000)] = 0.f;
  } else if (gid < 16850000) {
    fws[F_AGG + (gid - 15250000)] = 0.f;
  }
}

// ---------------------------------------------------------------------------
// Edge LSTM (K=64: [h32 | attr2 | pad], 128 gates) -> eh, ec, enc_sum, cnt
// ---------------------------------------------------------------------------
__global__ void __launch_bounds__(128) edge_lstm_kernel(
    const float* __restrict__ eattr, const float* __restrict__ heh,
    const float* __restrict__ hec, const int* __restrict__ eidx,
    const float* __restrict__ ebias, const _Float16* __restrict__ wp,
    float* __restrict__ out_eh, float* __restrict__ out_ec,
    float* __restrict__ enc_sum, float* __restrict__ cnt) {
  __shared__ alignas(32) _Float16 sW[64 * 128];
  __shared__ alignas(32) _Float16 ast[4][16 * 64];
  for (int i = threadIdx.x; i < 8192 / 8; i += 128)
    ((uint4*)sW)[i] = ((const uint4*)wp)[i];
  const int lane = threadIdx.x & 31;
  const int wave = threadIdx.x >> 5;
  const int e0 = (blockIdx.x * 4 + wave) * 16;
  _Float16* st = ast[wave];
  {
    const int m = lane >> 1, hf = lane & 1;
    const int e = e0 + m;
    const float4* hp = (const float4*)(heh + e * 32 + hf * 16);
    float4 h0 = hp[0], h1 = hp[1], h2 = hp[2], h3 = hp[3];
    *(v8h*)(st + m * 64 + hf * 16)     = cvt8(h0, h1);
    *(v8h*)(st + m * 64 + hf * 16 + 8) = cvt8(h2, h3);
    if (hf == 0) {
      v8h t = {};
      t[0] = (_Float16)eattr[e * 2 + 0];
      t[1] = (_Float16)eattr[e * 2 + 1];
      *(v8h*)(st + m * 64 + 32) = t;
    } else {
      v8h z = {};
      *(v8h*)(st + m * 64 + 40) = z;
      *(v8h*)(st + m * 64 + 48) = z;
      *(v8h*)(st + m * 64 + 56) = z;
    }
  }
  __syncthreads();
  v8f acc[8] = {};
#pragma unroll
  for (int kt = 0; kt < 2; ++kt) {
    v16h a = load_a(st, 64, lane, kt * 32);
#pragma unroll
    for (int t = 0; t < 8; ++t)
      acc[t] = wmma32(a, load_b(sW, 8, kt, t, lane), acc[t]);
  }
  const int nlo = lane & 15;
  const int radd = (lane >> 4) * 8;
  float bias[8];
#pragma unroll
  for (int t = 0; t < 8; ++t) bias[t] = ebias[t * 16 + nlo];
#pragma unroll
  for (int r = 0; r < 8; ++r) {
    const int e = e0 + r + radd;
    const int d = eidx[NE + e];
#pragma unroll
    for (int hh = 0; hh < 2; ++hh) {
      const int j = nlo + hh * 16;
      const float gi = acc[0 + hh][r] + bias[0 + hh];
      const float gf = acc[2 + hh][r] + bias[2 + hh];
      const float gg = acc[4 + hh][r] + bias[4 + hh];
      const float go = acc[6 + hh][r] + bias[6 + hh];
      const float cp = hec[e * 32 + j];
      const float c2 = sigmf_(gf) * cp + sigmf_(gi) * tanhf(gg);
      const float h2 = sigmf_(go) * tanhf(c2);
      out_eh[e * 32 + j] = h2;
      out_ec[e * 32 + j] = c2;
      atomicAdd(&enc_sum[d * 32 + j], h2);
    }
    if (nlo == 0) atomicAdd(&cnt[d], 1.0f);
  }
}

// ---------------------------------------------------------------------------
// Node LSTM (K=96: [h64 | x5 | pad], 256 gates) -> nh, nc
// ---------------------------------------------------------------------------
__global__ void __launch_bounds__(128) node_lstm_kernel(
    const float* __restrict__ x, const float* __restrict__ hnh,
    const float* __restrict__ hnc, const float* __restrict__ nbias,
    const _Float16* __restrict__ wp, float* __restrict__ out_nh,
    float* __restrict__ out_nc) {
  __shared__ alignas(32) _Float16 sW[96 * 256];
  __shared__ alignas(32) _Float16 ast[4][16 * 96];
  for (int i = threadIdx.x; i < 24576 / 8; i += 128)
    ((uint4*)sW)[i] = ((const uint4*)wp)[i];
  const int lane = threadIdx.x & 31;
  const int wave = threadIdx.x >> 5;
  const int n0 = (blockIdx.x * 4 + wave) * 16;
  _Float16* st = ast[wave];
  {
    const int m = lane >> 1, hf = lane & 1;
    const int n = n0 + m;
    v8h z = {};
    if (n < NN) {
      const float4* hp = (const float4*)(hnh + n * 64 + hf * 32);
      float4 a0 = hp[0], a1 = hp[1], a2 = hp[2], a3 = hp[3];
      float4 a4 = hp[4], a5 = hp[5], a6 = hp[6], a7 = hp[7];
      *(v8h*)(st + m * 96 + hf * 32 + 0)  = cvt8(a0, a1);
      *(v8h*)(st + m * 96 + hf * 32 + 8)  = cvt8(a2, a3);
      *(v8h*)(st + m * 96 + hf * 32 + 16) = cvt8(a4, a5);
      *(v8h*)(st + m * 96 + hf * 32 + 24) = cvt8(a6, a7);
      if (hf == 0) {
        v8h t = {};
        t[0] = (_Float16)x[n * 5 + 0]; t[1] = (_Float16)x[n * 5 + 1];
        t[2] = (_Float16)x[n * 5 + 2]; t[3] = (_Float16)x[n * 5 + 3];
        t[4] = (_Float16)x[n * 5 + 4];
        *(v8h*)(st + m * 96 + 64) = t;
      } else {
        *(v8h*)(st + m * 96 + 72) = z;
        *(v8h*)(st + m * 96 + 80) = z;
        *(v8h*)(st + m * 96 + 88) = z;
      }
    } else {
#pragma unroll
      for (int q = 0; q < 4; ++q) *(v8h*)(st + m * 96 + hf * 32 + q * 8) = z;
      if (hf == 0) *(v8h*)(st + m * 96 + 64) = z;
      else {
        *(v8h*)(st + m * 96 + 72) = z;
        *(v8h*)(st + m * 96 + 80) = z;
        *(v8h*)(st + m * 96 + 88) = z;
      }
    }
  }
  __syncthreads();
  v8f acc[16] = {};
#pragma unroll
  for (int kt = 0; kt < 3; ++kt) {
    v16h a = load_a(st, 96, lane, kt * 32);
#pragma unroll
    for (int t = 0; t < 16; ++t)
      acc[t] = wmma32(a, load_b(sW, 16, kt, t, lane), acc[t]);
  }
  const int nlo = lane & 15;
  const int radd = (lane >> 4) * 8;
  float bias[16];
#pragma unroll
  for (int t = 0; t < 16; ++t) bias[t] = nbias[t * 16 + nlo];
#pragma unroll
  for (int r = 0; r < 8; ++r) {
    const int n = n0 + r + radd;
    if (n >= NN) continue;
#pragma unroll
    for (int tt = 0; tt < 4; ++tt) {
      const int j = nlo + tt * 16;
      const float gi = acc[tt][r]      + bias[tt];
      const float gf = acc[4 + tt][r]  + bias[4 + tt];
      const float gg = acc[8 + tt][r]  + bias[8 + tt];
      const float go = acc[12 + tt][r] + bias[12 + tt];
      const float cp = hnc[n * 64 + j];
      const float c2 = sigmf_(gf) * cp + sigmf_(gi) * tanhf(gg);
      const float h2 = sigmf_(go) * tanhf(c2);
      out_nh[n * 64 + j] = h2;
      out_nc[n * 64 + j] = c2;
    }
  }
}

__global__ void enc_fin_kernel(float* __restrict__ enc, const float* __restrict__ cnt) {
  const int gid = blockIdx.x * 256 + threadIdx.x;
  if (gid >= NN * 32) return;
  enc[gid] /= fmaxf(cnt[gid >> 5], 1.f);
}

// ---------------------------------------------------------------------------
// GAT projection (K=96: [nh64 | enc32], 256 out) -> xh
// ---------------------------------------------------------------------------
__global__ void __launch_bounds__(128) gat_proj_kernel(
    const float* __restrict__ nh, const float* __restrict__ enc,
    const _Float16* __restrict__ wp, float* __restrict__ xh) {
  __shared__ alignas(32) _Float16 sW[96 * 256];
  __shared__ alignas(32) _Float16 ast[4][16 * 96];
  for (int i = threadIdx.x; i < 24576 / 8; i += 128)
    ((uint4*)sW)[i] = ((const uint4*)wp)[i];
  const int lane = threadIdx.x & 31;
  const int wave = threadIdx.x >> 5;
  const int n0 = (blockIdx.x * 4 + wave) * 16;
  _Float16* st = ast[wave];
  {
    const int m = lane >> 1, hf = lane & 1;
    const int n = n0 + m;
    v8h z = {};
    if (n < NN) {
      const float4* hp = (const float4*)(nh + n * 64 + hf * 32);
      float4 a0 = hp[0], a1 = hp[1], a2 = hp[2], a3 = hp[3];
      float4 a4 = hp[4], a5 = hp[5], a6 = hp[6], a7 = hp[7];
      *(v8h*)(st + m * 96 + hf * 32 + 0)  = cvt8(a0, a1);
      *(v8h*)(st + m * 96 + hf * 32 + 8)  = cvt8(a2, a3);
      *(v8h*)(st + m * 96 + hf * 32 + 16) = cvt8(a4, a5);
      *(v8h*)(st + m * 96 + hf * 32 + 24) = cvt8(a6, a7);
      const float4* ep = (const float4*)(enc + n * 32 + hf * 16);
      float4 e0v = ep[0], e1v = ep[1], e2v = ep[2], e3v = ep[3];
      *(v8h*)(st + m * 96 + 64 + hf * 16)     = cvt8(e0v, e1v);
      *(v8h*)(st + m * 96 + 64 + hf * 16 + 8) = cvt8(e2v, e3v);
    } else {
#pragma unroll
      for (int q = 0; q < 4; ++q) *(v8h*)(st + m * 96 + hf * 32 + q * 8) = z;
      *(v8h*)(st + m * 96 + 64 + hf * 16)     = z;
      *(v8h*)(st + m * 96 + 64 + hf * 16 + 8) = z;
    }
  }
  __syncthreads();
  v8f acc[16] = {};
#pragma unroll
  for (int kt = 0; kt < 3; ++kt) {
    v16h a = load_a(st, 96, lane, kt * 32);
#pragma unroll
    for (int t = 0; t < 16; ++t)
      acc[t] = wmma32(a, load_b(sW, 16, kt, t, lane), acc[t]);
  }
  const int nlo = lane & 15;
  const int radd = (lane >> 4) * 8;
#pragma unroll
  for (int r = 0; r < 8; ++r) {
    const int n = n0 + r + radd;
    if (n >= NN) continue;
#pragma unroll
    for (int t = 0; t < 16; ++t) xh[n * 256 + t * 16 + nlo] = acc[t][r];
  }
}

__global__ void att_coef_kernel(const float* __restrict__ xh,
                                const float* __restrict__ attS,
                                const float* __restrict__ attD,
                                float* __restrict__ as_, float* __restrict__ ad_) {
  const int gid = blockIdx.x * 256 + threadIdx.x;
  if (gid >= NN * 4) return;
  const int n = gid >> 2, h = gid & 3;
  float s = 0.f, d = 0.f;
  for (int k = 0; k < 64; ++k) {
    const float v = xh[n * 256 + h * 64 + k];
    s += v * attS[h * 64 + k];
    d += v * attD[h * 64 + k];
  }
  as_[gid] = s;
  ad_[gid] = d;
}

__global__ void att_max_kernel(const int* __restrict__ eidx,
                               const float* __restrict__ as_,
                               const float* __restrict__ ad_,
                               float* __restrict__ mmax) {
  const int gid = blockIdx.x * 256 + threadIdx.x;
  if (gid >= (NE + NN) * 4) return;
  const int eid = gid >> 2, h = gid & 3;
  int s, d;
  if (eid < NE) { s = eidx[eid]; d = eidx[NE + eid]; } else { s = d = eid - NE; }
  atomic_max_f(&mmax[d * 4 + h], lrelu_(as_[s * 4 + h] + ad_[d * 4 + h]));
}

__global__ void att_denom_kernel(const int* __restrict__ eidx,
                                 const float* __restrict__ as_,
                                 const float* __restrict__ ad_,
                                 const float* __restrict__ mmax,
                                 float* __restrict__ denom) {
  const int gid = blockIdx.x * 256 + threadIdx.x;
  if (gid >= (NE + NN) * 4) return;
  const int eid = gid >> 2, h = gid & 3;
  int s, d;
  if (eid < NE) { s = eidx[eid]; d = eidx[NE + eid]; } else { s = d = eid - NE; }
  const float v = lrelu_(as_[s * 4 + h] + ad_[d * 4 + h]);
  atomicAdd(&denom[d * 4 + h], __expf(v - mmax[d * 4 + h]));
}

__global__ void gat_scatter_kernel(const int* __restrict__ eidx,
                                   const float* __restrict__ as_,
                                   const float* __restrict__ ad_,
                                   const float* __restrict__ mmax,
                                   const float* __restrict__ xh,
                                   float* __restrict__ gsum) {
  const int gid = blockIdx.x * 256 + threadIdx.x;
  if (gid >= (NE + NN) * 64) return;
  const int eid = gid >> 6;
  const int c0 = (gid & 63) * 4;
  const int h = c0 >> 6;
  int s, d;
  if (eid < NE) { s = eidx[eid]; d = eidx[NE + eid]; } else { s = d = eid - NE; }
  const float v  = lrelu_(as_[s * 4 + h] + ad_[d * 4 + h]);
  const float al = __expf(v - mmax[d * 4 + h]);
#pragma unroll
  for (int j = 0; j < 4; ++j) {
    const int c = c0 + j;
    atomicAdd(&gsum[d * 256 + c], xh[s * 256 + c] * al);
  }
}

__global__ void xgat_kernel(const float* __restrict__ gsum,
                            const float* __restrict__ denom,
                            const float* __restrict__ gatB,
                            float* __restrict__ xgat) {
  const int gid = blockIdx.x * 256 + threadIdx.x;
  if (gid >= NN * 64) return;
  const int n = gid >> 6, dd = gid & 63;
  float s = 0.f;
#pragma unroll
  for (int h = 0; h < 4; ++h) s += gsum[n * 256 + h * 64 + dd] / denom[n * 4 + h];
  xgat[gid] = s * 0.25f + gatB[dd];
}

// ---------------------------------------------------------------------------
// GN edge MLP (K=160: [xg[src]64 | xg[dst]64 | attr2 | pad]) -> agg[dst]
// ---------------------------------------------------------------------------
__global__ void __launch_bounds__(128) edge_mlp_kernel(
    const float* __restrict__ xgat, const float* __restrict__ eattr,
    const int* __restrict__ eidx, const float* __restrict__ b1,
    const float* __restrict__ b2, const _Float16* __restrict__ wp1,
    const _Float16* __restrict__ wp2, float* __restrict__ agg) {
  __shared__ alignas(32) _Float16 sW1[160 * 64];
  __shared__ alignas(32) _Float16 sW2[64 * 32];
  __shared__ alignas(32) _Float16 ast[4][16 * 160];
  for (int i = threadIdx.x; i < 10240 / 8; i += 128)
    ((uint4*)sW1)[i] = ((const uint4*)wp1)[i];
  for (int i = threadIdx.x; i < 2048 / 8; i += 128)
    ((uint4*)sW2)[i] = ((const uint4*)wp2)[i];
  const int lane = threadIdx.x & 31;
  const int wave = threadIdx.x >> 5;
  const int e0 = (blockIdx.x * 4 + wave) * 16;
  _Float16* st = ast[wave];
  {
    const int m = lane >> 1, hf = lane & 1;
    const int e = e0 + m;
    const int s = eidx[e], d = eidx[NE + e];
    const float4* sp = (const float4*)(xgat + s * 64 + hf * 32);
    const float4* dp = (const float4*)(xgat + d * 64 + hf * 32);
    float4 s0 = sp[0], s1 = sp[1], s2 = sp[2], s3 = sp[3];
    float4 s4 = sp[4], s5 = sp[5], s6 = sp[6], s7 = sp[7];
    float4 d0 = dp[0], d1 = dp[1], d2 = dp[2], d3 = dp[3];
    float4 d4 = dp[4], d5 = dp[5], d6 = dp[6], d7 = dp[7];
    *(v8h*)(st + m * 160 + hf * 32 + 0)  = cvt8(s0, s1);
    *(v8h*)(st + m * 160 + hf * 32 + 8)  = cvt8(s2, s3);
    *(v8h*)(st + m * 160 + hf * 32 + 16) = cvt8(s4, s5);
    *(v8h*)(st + m * 160 + hf * 32 + 24) = cvt8(s6, s7);
    *(v8h*)(st + m * 160 + 64 + hf * 32 + 0)  = cvt8(d0, d1);
    *(v8h*)(st + m * 160 + 64 + hf * 32 + 8)  = cvt8(d2, d3);
    *(v8h*)(st + m * 160 + 64 + hf * 32 + 16) = cvt8(d4, d5);
    *(v8h*)(st + m * 160 + 64 + hf * 32 + 24) = cvt8(d6, d7);
    if (hf == 0) {
      v8h t = {};
      t[0] = (_Float16)eattr[e * 2 + 0];
      t[1] = (_Float16)eattr[e * 2 + 1];
      *(v8h*)(st + m * 160 + 128) = t;
    } else {
      v8h z = {};
      *(v8h*)(st + m * 160 + 136) = z;
      *(v8h*)(st + m * 160 + 144) = z;
      *(v8h*)(st + m * 160 + 152) = z;
    }
  }
  __syncthreads();
  v8f acc1[4] = {};
#pragma unroll
  for (int kt = 0; kt < 5; ++kt) {
    v16h a = load_a(st, 160, lane, kt * 32);
#pragma unroll
    for (int t = 0; t < 4; ++t)
      acc1[t] = wmma32(a, load_b(sW1, 4, kt, t, lane), acc1[t]);
  }
  const int nlo = lane & 15;
  const int radd = (lane >> 4) * 8;
  // transpose relu(hidden) through per-wave LDS (reuse staging buffer)
#pragma unroll
  for (int t = 0; t < 4; ++t)
#pragma unroll
    for (int r = 0; r < 8; ++r)
      st[(r + radd) * 64 + t * 16 + nlo] =
          (_Float16)fmaxf(acc1[t][r] + b1[t * 16 + nlo], 0.f);
  v8f acc2[2] = {};
#pragma unroll
  for (int kt = 0; kt < 2; ++kt) {
    v16h a = load_a(st, 64, lane, kt * 32);
#pragma unroll
    for (int t = 0; t < 2; ++t)
      acc2[t] = wmma32(a, load_b(sW2, 2, kt, t, lane), acc2[t]);
  }
#pragma unroll
  for (int r = 0; r < 8; ++r) {
    const int e = e0 + r + radd;
    const int d = eidx[NE + e];
#pragma unroll
    for (int t = 0; t < 2; ++t) {
      const int j = t * 16 + nlo;
      atomicAdd(&agg[d * 32 + j], acc2[t][r] + b2[j]);
    }
  }
}

// ---------------------------------------------------------------------------
// GN node MLP (K=96: [xg64 | agg/cnt 32]) -> out (N x 4)
// ---------------------------------------------------------------------------
__global__ void __launch_bounds__(128) node_mlp_kernel(
    const float* __restrict__ xgat, const float* __restrict__ agg,
    const float* __restrict__ cnt, const float* __restrict__ b1,
    const float* __restrict__ b2, const _Float16* __restrict__ wp1,
    const _Float16* __restrict__ wp2, float* __restrict__ out) {
  __shared__ alignas(32) _Float16 sW1[96 * 64];
  __shared__ alignas(32) _Float16 sW2[64 * 16];
  __shared__ alignas(32) _Float16 ast[4][16 * 96];
  for (int i = threadIdx.x; i < 6144 / 8; i += 128)
    ((uint4*)sW1)[i] = ((const uint4*)wp1)[i];
  for (int i = threadIdx.x; i < 1024 / 8; i += 128)
    ((uint4*)sW2)[i] = ((const uint4*)wp2)[i];
  const int lane = threadIdx.x & 31;
  const int wave = threadIdx.x >> 5;
  const int n0 = (blockIdx.x * 4 + wave) * 16;
  _Float16* st = ast[wave];
  {
    const int m = lane >> 1, hf = lane & 1;
    const int n = n0 + m;
    v8h z = {};
    if (n < NN) {
      const float inv = 1.f / fmaxf(cnt[n], 1.f);
      const float4* gp = (const float4*)(xgat + n * 64 + hf * 32);
      float4 a0 = gp[0], a1 = gp[1], a2 = gp[2], a3 = gp[3];
      float4 a4 = gp[4], a5 = gp[5], a6 = gp[6], a7 = gp[7];
      *(v8h*)(st + m * 96 + hf * 32 + 0)  = cvt8(a0, a1);
      *(v8h*)(st + m * 96 + hf * 32 + 8)  = cvt8(a2, a3);
      *(v8h*)(st + m * 96 + hf * 32 + 16) = cvt8(a4, a5);
      *(v8h*)(st + m * 96 + hf * 32 + 24) = cvt8(a6, a7);
      const float4* ap = (const float4*)(agg + n * 32 + hf * 16);
      float4 g0 = ap[0], g1 = ap[1], g2 = ap[2], g3 = ap[3];
      g0.x *= inv; g0.y *= inv; g0.z *= inv; g0.w *= inv;
      g1.x *= inv; g1.y *= inv; g1.z *= inv; g1.w *= inv;
      g2.x *= inv; g2.y *= inv; g2.z *= inv; g2.w *= inv;
      g3.x *= inv; g3.y *= inv; g3.z *= inv; g3.w *= inv;
      *(v8h*)(st + m * 96 + 64 + hf * 16)     = cvt8(g0, g1);
      *(v8h*)(st + m * 96 + 64 + hf * 16 + 8) = cvt8(g2, g3);
    } else {
#pragma unroll
      for (int q = 0; q < 4; ++q) *(v8h*)(st + m * 96 + hf * 32 + q * 8) = z;
      *(v8h*)(st + m * 96 + 64 + hf * 16)     = z;
      *(v8h*)(st + m * 96 + 64 + hf * 16 + 8) = z;
    }
  }
  __syncthreads();
  v8f acc1[4] = {};
#pragma unroll
  for (int kt = 0; kt < 3; ++kt) {
    v16h a = load_a(st, 96, lane, kt * 32);
#pragma unroll
    for (int t = 0; t < 4; ++t)
      acc1[t] = wmma32(a, load_b(sW1, 4, kt, t, lane), acc1[t]);
  }
  const int nlo = lane & 15;
  const int radd = (lane >> 4) * 8;
#pragma unroll
  for (int t = 0; t < 4; ++t)
#pragma unroll
    for (int r = 0; r < 8; ++r)
      st[(r + radd) * 64 + t * 16 + nlo] =
          (_Float16)fmaxf(acc1[t][r] + b1[t * 16 + nlo], 0.f);
  v8f acc2 = {};
#pragma unroll
  for (int kt = 0; kt < 2; ++kt) {
    v16h a = load_a(st, 64, lane, kt * 32);
    acc2 = wmma32(a, load_b(sW2, 1, kt, 0, lane), acc2);
  }
  if (nlo < 4) {
#pragma unroll
    for (int r = 0; r < 8; ++r) {
      const int n = n0 + r + radd;
      if (n < NN) out[n * 4 + nlo] = acc2[r] + b2[nlo];
    }
  }
}

// ---------------------------------------------------------------------------
// Launch
// ---------------------------------------------------------------------------
extern "C" void kernel_launch(void* const* d_in, const int* in_sizes, int n_in,
                              void* d_out, int out_size, void* d_ws, size_t ws_size,
                              hipStream_t stream) {
  const float* x     = (const float*)d_in[0];
  const int*   eidx  = (const int*)d_in[1];
  const float* eattr = (const float*)d_in[2];
  const float* hnh   = (const float*)d_in[3];
  const float* hnc   = (const float*)d_in[4];
  const float* heh   = (const float*)d_in[5];
  const float* hec   = (const float*)d_in[6];
  const float* nWih  = (const float*)d_in[7];
  const float* nWhh  = (const float*)d_in[8];
  const float* nb    = (const float*)d_in[9];
  const float* eWih  = (const float*)d_in[10];
  const float* eWhh  = (const float*)d_in[11];
  const float* eb    = (const float*)d_in[12];
  const float* gatW  = (const float*)d_in[13];
  const float* attS  = (const float*)d_in[14];
  const float* attD  = (const float*)d_in[15];
  const float* gatB  = (const float*)d_in[16];
  const float* emW1  = (const float*)d_in[17];
  const float* emb1  = (const float*)d_in[18];
  const float* emW2  = (const float*)d_in[19];
  const float* emb2  = (const float*)d_in[20];
  const float* nmW1  = (const float*)d_in[21];
  const float* nmb1  = (const float*)d_in[22];
  const float* nmW2  = (const float*)d_in[23];
  const float* nmb2  = (const float*)d_in[24];

  float* out = (float*)d_out;
  char*  ws  = (char*)d_ws;
  float* fws = (float*)(ws + FWS_OFF);
  const _Float16* wpE  = (const _Float16*)(ws + WP_EDGE);
  const _Float16* wpN  = (const _Float16*)(ws + WP_NODE);
  const _Float16* wpG  = (const _Float16*)(ws + WP_GAT);
  const _Float16* wpM1 = (const _Float16*)(ws + WP_EM1);
  const _Float16* wpM2 = (const _Float16*)(ws + WP_EM2);
  const _Float16* wpQ1 = (const _Float16*)(ws + WP_NM1);
  const _Float16* wpQ2 = (const _Float16*)(ws + WP_NM2);

  pack_weights_kernel<<<300, 256, 0, stream>>>(nWih, nWhh, eWih, eWhh, gatW,
                                               emW1, emW2, nmW1, nmW2, ws);
  init_ws_kernel<<<(16850000 + 255) / 256, 256, 0, stream>>>(fws);

  edge_lstm_kernel<<<NE / 64, 128, 0, stream>>>(
      eattr, heh, hec, eidx, eb, wpE, out + O_EH, out + O_EC,
      fws + F_ENC, fws + F_CNT);
  node_lstm_kernel<<<(NN + 63) / 64, 128, 0, stream>>>(
      x, hnh, hnc, nb, wpN, out + O_NH, out + O_NC);
  enc_fin_kernel<<<(NN * 32 + 255) / 256, 256, 0, stream>>>(fws + F_ENC, fws + F_CNT);

  gat_proj_kernel<<<(NN + 63) / 64, 128, 0, stream>>>(
      out + O_NH, fws + F_ENC, wpG, fws + F_XH);
  att_coef_kernel<<<(NN * 4 + 255) / 256, 256, 0, stream>>>(
      fws + F_XH, attS, attD, fws + F_AS, fws + F_AD);
  att_max_kernel<<<((NE + NN) * 4 + 255) / 256, 256, 0, stream>>>(
      eidx, fws + F_AS, fws + F_AD, fws + F_MM);
  att_denom_kernel<<<((NE + NN) * 4 + 255) / 256, 256, 0, stream>>>(
      eidx, fws + F_AS, fws + F_AD, fws + F_MM, fws + F_DEN);
  gat_scatter_kernel<<<((NE + NN) * 64 + 255) / 256, 256, 0, stream>>>(
      eidx, fws + F_AS, fws + F_AD, fws + F_MM, fws + F_XH, fws + F_GS);
  xgat_kernel<<<(NN * 64 + 255) / 256, 256, 0, stream>>>(
      fws + F_GS, fws + F_DEN, gatB, fws + F_XG);

  edge_mlp_kernel<<<NE / 64, 128, 0, stream>>>(
      fws + F_XG, eattr, eidx, emb1, emb2, wpM1, wpM2, fws + F_AGG);
  node_mlp_kernel<<<(NN + 63) / 64, 128, 0, stream>>>(
      fws + F_XG, fws + F_AGG, fws + F_CNT, nmb1, nmb2, wpQ1, wpQ2, out + O_OUT);
}
